// SparseMultiHeadAttention_17557826306131
// MI455X (gfx1250) — compile-verified, matrix operand order checked by reading, not run on hardware
//
#include <hip/hip_runtime.h>

#define HIDDEN 768
#define NHEAD  12
#define HDIM   64
#define NNODES 32768
#define NEDGES 1048576
#define KTILES (HIDDEN / 32)   // 24

typedef __attribute__((ext_vector_type(16))) __bf16 v16bf;
typedef __attribute__((ext_vector_type(8)))  float  v8f;

union FragBF {
    v16bf v;
    unsigned short u[16];
    unsigned int   w[8];
    uint4          q[2];
};

__device__ __forceinline__ unsigned short f32_to_bf16(float x) {
    unsigned int u = __float_as_uint(x);
    unsigned int r = u + 0x7FFFu + ((u >> 16) & 1u);   // round-to-nearest-even
    return (unsigned short)(r >> 16);
}

__device__ __forceinline__ unsigned int pack_bf16x2(float a, float b) {
    return (unsigned int)f32_to_bf16(a) | ((unsigned int)f32_to_bf16(b) << 16);
}

// Fragment-order addressing for a 16-wide subtile, K-tile of 32 (CDNA5 bf16 WMMA):
//   lane = (row_or_col % 16) + 16*lhi,  lhi = (klocal>>3)&1
//   elem j = (klocal & 7) + 8*(klocal>>4)
// One subtile/K-tile fragment block = 32 lanes * 16 bf16 = 1 KB contiguous.
__device__ __forceinline__ size_t frag_idx(int tileIdx, int lo, int klocal) {
    int lhi  = (klocal >> 3) & 1;
    int j    = (klocal & 7) + ((klocal >> 4) << 3);
    int lane = lo + (lhi << 4);
    return ((size_t)tileIdx << 9) + lane * 16 + j;
}

// ---------------------------------------------------------------------------
// Weights -> bf16, transposed to B[k][n] = W[n][k], stored in fragment order:
//   block index = kt * (ncols/16) + (n/16)
// ---------------------------------------------------------------------------
__global__ __launch_bounds__(256)
void convert_weights_kernel(const float* __restrict__ Wq, const float* __restrict__ Wk,
                            const float* __restrict__ Wv, const float* __restrict__ Wo,
                            unsigned short* __restrict__ Wq_sw,
                            unsigned short* __restrict__ Wkv_sw,
                            unsigned short* __restrict__ Wo_sw)
{
    int t = blockIdx.x * 256 + threadIdx.x;    // 3072 * 384 threads
    int n = t / (HIDDEN / 2);                  // 0..3071
    int k = (t % (HIDDEN / 2)) * 2;            // even k, pair (k, k+1)

    const float* srcRow;
    unsigned short* dstB;
    int col, nt16;
    if (n < HIDDEN)            { srcRow = Wq + (size_t)n * HIDDEN;            dstB = Wq_sw;  col = n;           nt16 = HIDDEN / 16; }
    else if (n < 2 * HIDDEN)   { srcRow = Wk + (size_t)(n - HIDDEN) * HIDDEN; dstB = Wkv_sw; col = n - HIDDEN;  nt16 = 2 * HIDDEN / 16; }
    else if (n < 3 * HIDDEN)   { srcRow = Wv + (size_t)(n - 2 * HIDDEN) * HIDDEN; dstB = Wkv_sw; col = n - HIDDEN; nt16 = 2 * HIDDEN / 16; }
    else                       { srcRow = Wo + (size_t)(n - 3 * HIDDEN) * HIDDEN; dstB = Wo_sw;  col = n - 3 * HIDDEN; nt16 = HIDDEN / 16; }

    float2 xv = *(const float2*)(srcRow + k);
    int kt = k >> 5, kl = k & 31;
    size_t idx = frag_idx(kt * nt16 + (col >> 4), col & 15, kl);   // kl even -> j even
    *(unsigned int*)(dstB + idx) = pack_bf16x2(xv.x, xv.y);
}

// ---------------------------------------------------------------------------
// Activations f32 [M][768] -> bf16 fragment order:
//   block index = (m/16) * KTILES + kt   (K-tiles contiguous per row-subtile)
// ---------------------------------------------------------------------------
__global__ __launch_bounds__(256)
void convert_act_kernel(const float* __restrict__ src, unsigned short* __restrict__ dst)
{
    int t = blockIdx.x * 256 + threadIdx.x;    // NNODES * 384 threads
    int m = t / (HIDDEN / 2);
    int k = (t % (HIDDEN / 2)) * 2;
    float2 xv = *(const float2*)(src + (size_t)m * HIDDEN + k);
    int kt = k >> 5, kl = k & 31;
    size_t idx = frag_idx((m >> 4) * KTILES + kt, m & 15, kl);
    *(unsigned int*)(dst + idx) = pack_bf16x2(xv.x, xv.y);
}

// ---------------------------------------------------------------------------
// CSR row pointers from sorted row_idx (lower_bound per row).
// ---------------------------------------------------------------------------
__global__ __launch_bounds__(256)
void rowptr_kernel(const int* __restrict__ row_idx, int* __restrict__ row_ptr)
{
    int t = blockIdx.x * 256 + threadIdx.x;
    if (t > NNODES) return;
    int lo = 0, hi = NEDGES;
    while (lo < hi) {
        int mid = (lo + hi) >> 1;
        if (row_idx[mid] < t) lo = mid + 1; else hi = mid;
    }
    row_ptr[t] = lo;
}

// ---------------------------------------------------------------------------
// WMMA GEMM, no LDS: both operands pre-swizzled to fragment order in global.
// Block = 128x128 tile, 8 waves; wave = 32x64 (2x4 subtiles of 16x16).
// Per K-step per wave: 12x global_load_b128 feeding 8x v_wmma (bf16, f32 acc).
// mode 0: Q   -> out0[m][h*64+d] = (C + bias0[f]) * 1/8,  f = d*12+h
// mode 1: K|V -> left 768 cols -> out0 (+bias0), right -> out1 (+bias1), permuted
// mode 2: O   -> out0[m][n] = C + bias0[n]   (row-major, to d_out)
// ---------------------------------------------------------------------------
__global__ __launch_bounds__(256)
void wmma_gemm_kernel(const unsigned short* __restrict__ A_sw,
                      const unsigned short* __restrict__ B_sw, int nt16,
                      const float* __restrict__ bias0, const float* __restrict__ bias1,
                      float* __restrict__ out0, float* __restrict__ out1,
                      int mode)
{
    const int lane = threadIdx.x & 31;
    const int w    = threadIdx.x >> 5;
    const int wm   = w & 3;          // M position (x32)
    const int wn   = w >> 2;         // N position (x64)
    const int llo  = lane & 15;
    const int lhi  = lane >> 4;

    const int mtile0 = blockIdx.y * 8 + wm * 2;   // first of 2 M-subtiles
    const int ntile0 = blockIdx.x * 8 + wn * 4;   // first of 4 N-subtiles

    const unsigned short* aP[2];
    const unsigned short* bP[4];
#pragma unroll
    for (int sm = 0; sm < 2; ++sm)
        aP[sm] = A_sw + ((size_t)(mtile0 + sm) * KTILES << 9) + lane * 16;
#pragma unroll
    for (int sn = 0; sn < 4; ++sn)
        bP[sn] = B_sw + ((size_t)(ntile0 + sn) << 9) + lane * 16;

    const size_t bStep = (size_t)nt16 << 9;       // B advance per K-tile

    v8f acc[2][4] = {};

#pragma unroll 2
    for (int kt = 0; kt < KTILES; ++kt) {
        FragBF a[2], b[4];
#pragma unroll
        for (int sm = 0; sm < 2; ++sm) {
            const uint4* p = (const uint4*)(aP[sm] + (size_t)kt * 512);
            a[sm].q[0] = p[0];
            a[sm].q[1] = p[1];
        }
#pragma unroll
        for (int sn = 0; sn < 4; ++sn) {
            const uint4* p = (const uint4*)(bP[sn] + (size_t)kt * bStep);
            b[sn].q[0] = p[0];
            b[sn].q[1] = p[1];
        }
#pragma unroll
        for (int sm = 0; sm < 2; ++sm)
#pragma unroll
            for (int sn = 0; sn < 4; ++sn)
                acc[sm][sn] = __builtin_amdgcn_wmma_f32_16x16x32_bf16(
                    false, a[sm].v, false, b[sn].v,
                    (short)0, acc[sm][sn], false, false);
    }

    // C/D layout: VGPR i holds M = i + 8*lhi, N = llo (within 16x16 subtile)
#pragma unroll
    for (int sm = 0; sm < 2; ++sm) {
#pragma unroll
        for (int sn = 0; sn < 4; ++sn) {
            int gn = (ntile0 + sn) * 16 + llo;
#pragma unroll
            for (int i = 0; i < 8; ++i) {
                int gm = (mtile0 + sm) * 16 + lhi * 8 + i;
                float val = acc[sm][sn][i];
                if (mode == 0) {                        // Q: bias, scale, permute
                    int d = gn / NHEAD, h = gn % NHEAD;
                    val = (val + bias0[gn]) * 0.125f;
                    out0[(size_t)gm * HIDDEN + h * HDIM + d] = val;
                } else if (mode == 1) {                 // K or V: bias, permute
                    int sel = gn / HIDDEN, f = gn % HIDDEN;
                    int d = f / NHEAD, h = f % NHEAD;
                    val += sel ? bias1[f] : bias0[f];
                    float* o = sel ? out1 : out0;
                    o[(size_t)gm * HIDDEN + h * HDIM + d] = val;
                } else {                                // O projection
                    out0[(size_t)gm * HIDDEN + gn] = val + bias0[gn];
                }
            }
        }
    }
}

// ---------------------------------------------------------------------------
// SDDMM: scores[e][h] = dot64(q[row[e]][h][:], k[col[e]][h][:])
// one thread per (edge, head); q/k rows are L2-resident (100 MB each).
// ---------------------------------------------------------------------------
__global__ __launch_bounds__(256)
void sddmm_kernel(const int* __restrict__ row_idx, const int* __restrict__ col_idx,
                  const float* __restrict__ qbuf, const float* __restrict__ kbuf,
                  float* __restrict__ scores)
{
    long long t = (long long)blockIdx.x * 256 + threadIdx.x;
    if (t >= (long long)NEDGES * NHEAD) return;
    int e = (int)(t / NHEAD);
    int h = (int)(t % NHEAD);
    int r = row_idx[e], c = col_idx[e];
    const float4* qp = (const float4*)(qbuf + (size_t)r * HIDDEN + h * HDIM);
    const float4* kp = (const float4*)(kbuf + (size_t)c * HIDDEN + h * HDIM);
    float acc = 0.f;
#pragma unroll
    for (int j = 0; j < 16; ++j) {
        float4 a = qp[j], b = kp[j];
        acc += a.x * b.x + a.y * b.y + a.z * b.z + a.w * b.w;
    }
    scores[(size_t)e * NHEAD + h] = acc;
}

// ---------------------------------------------------------------------------
// Segment softmax per (target row, head), in place over the row's edges.
// ---------------------------------------------------------------------------
__global__ __launch_bounds__(256)
void softmax_kernel(const int* __restrict__ row_ptr, float* __restrict__ scores)
{
    int t = blockIdx.x * 256 + threadIdx.x;
    if (t >= NNODES * NHEAD) return;
    int row = t / NHEAD, h = t % NHEAD;
    int s = row_ptr[row], epos = row_ptr[row + 1];
    if (s >= epos) return;
    float m = -3.4e38f;
    for (int e = s; e < epos; ++e)
        m = fmaxf(m, scores[(size_t)e * NHEAD + h]);
    float sum = 0.f;
    for (int e = s; e < epos; ++e) {
        float p = __expf(scores[(size_t)e * NHEAD + h] - m);
        scores[(size_t)e * NHEAD + h] = p;
        sum += p;
    }
    float inv = 1.f / sum;
    for (int e = s; e < epos; ++e)
        scores[(size_t)e * NHEAD + h] *= inv;
}

// ---------------------------------------------------------------------------
// SPMM: one wave per (row, head); 2 dims per lane; V rows from L2.
// Writes the result DIRECTLY in bf16 fragment order (f = d*12+h is the
// K index of the O-projection GEMM), so no extra conversion pass is needed.
// ---------------------------------------------------------------------------
__global__ __launch_bounds__(256)
void spmm_kernel(const int* __restrict__ row_ptr, const int* __restrict__ col_idx,
                 const float* __restrict__ attn, const float* __restrict__ vbuf,
                 unsigned short* __restrict__ osw)
{
    int gw = blockIdx.x * 8 + (threadIdx.x >> 5);
    if (gw >= NNODES * NHEAD) return;
    int lane = threadIdx.x & 31;
    int row = gw / NHEAD, h = gw % NHEAD;
    int s = row_ptr[row], epos = row_ptr[row + 1];
    int d = lane * 2;
    float a0 = 0.f, a1 = 0.f;
    for (int e = s; e < epos; ++e) {
        float wgt = attn[(size_t)e * NHEAD + h];
        int c = col_idx[e];
        float2 vv = *(const float2*)(vbuf + (size_t)c * HIDDEN + h * HDIM + d);
        a0 += wgt * vv.x;
        a1 += wgt * vv.y;
    }
    int f0 = d * NHEAD + h;
    int f1 = (d + 1) * NHEAD + h;
    int mt = row >> 4, mlo = row & 15;
    osw[frag_idx(mt * KTILES + (f0 >> 5), mlo, f0 & 31)] = f32_to_bf16(a0);
    osw[frag_idx(mt * KTILES + (f1 >> 5), mlo, f1 & 31)] = f32_to_bf16(a1);
}

// ---------------------------------------------------------------------------
extern "C" void kernel_launch(void* const* d_in, const int* in_sizes, int n_in,
                              void* d_out, int out_size, void* d_ws, size_t ws_size,
                              hipStream_t stream)
{
    const float* h_source = (const float*)d_in[0];
    const float* h_target = (const float*)d_in[1];
    const int*   row_idx  = (const int*)d_in[2];
    const int*   col_idx  = (const int*)d_in[3];
    const float* Wq = (const float*)d_in[4];
    const float* bq = (const float*)d_in[5];
    const float* Wk = (const float*)d_in[6];
    const float* bk = (const float*)d_in[7];
    const float* Wv = (const float*)d_in[8];
    const float* bv = (const float*)d_in[9];
    const float* Wo = (const float*)d_in[10];
    const float* bo = (const float*)d_in[11];
    float* out = (float*)d_out;
    (void)in_sizes; (void)n_in; (void)out_size; (void)ws_size;

    char* ws = (char*)d_ws;
    size_t off = 0;
    auto take = [&](size_t bytes) -> char* {
        char* p = ws + off;
        off = (off + bytes + 255) & ~(size_t)255;
        return p;
    };
    unsigned short* Wq_sw  = (unsigned short*)take((size_t)HIDDEN * HIDDEN * 2);
    unsigned short* Wkv_sw = (unsigned short*)take((size_t)HIDDEN * 2 * HIDDEN * 2);
    unsigned short* Wo_sw  = (unsigned short*)take((size_t)HIDDEN * HIDDEN * 2);
    unsigned short* At_sw  = (unsigned short*)take((size_t)NNODES * HIDDEN * 2);
    unsigned short* As_sw  = (unsigned short*)take((size_t)NNODES * HIDDEN * 2);
    unsigned short* osw    = (unsigned short*)take((size_t)NNODES * HIDDEN * 2);
    float* qbuf   = (float*)take((size_t)NNODES * HIDDEN * 4);
    float* kbuf   = (float*)take((size_t)NNODES * HIDDEN * 4);
    float* vbuf   = (float*)take((size_t)NNODES * HIDDEN * 4);
    float* scores = (float*)take((size_t)NEDGES * NHEAD * 4);
    int*   row_ptr = (int*)take((size_t)(NNODES + 1) * 4);

    // pre-passes
    convert_weights_kernel<<<(3072 * (HIDDEN / 2)) / 256, 256, 0, stream>>>(
        Wq, Wk, Wv, Wo, Wq_sw, Wkv_sw, Wo_sw);
    convert_act_kernel<<<(NNODES * (HIDDEN / 2)) / 256, 256, 0, stream>>>(h_target, At_sw);
    convert_act_kernel<<<(NNODES * (HIDDEN / 2)) / 256, 256, 0, stream>>>(h_source, As_sw);
    rowptr_kernel<<<(NNODES + 256) / 256, 256, 0, stream>>>(row_idx, row_ptr);

    // Q = h_target @ Wq^T  (mode 0)
    wmma_gemm_kernel<<<dim3(HIDDEN / 128, NNODES / 128), 256, 0, stream>>>(
        At_sw, Wq_sw, HIDDEN / 16, bq, nullptr, qbuf, nullptr, 0);
    // [K | V] = h_source @ [Wk | Wv]^T fused (mode 1)
    wmma_gemm_kernel<<<dim3(2 * HIDDEN / 128, NNODES / 128), 256, 0, stream>>>(
        As_sw, Wkv_sw, 2 * HIDDEN / 16, bk, bv, kbuf, vbuf, 1);

    sddmm_kernel<<<(NEDGES * NHEAD) / 256, 256, 0, stream>>>(
        row_idx, col_idx, qbuf, kbuf, scores);
    softmax_kernel<<<(NNODES * NHEAD) / 256, 256, 0, stream>>>(row_ptr, scores);
    spmm_kernel<<<(NNODES * NHEAD) / 8, 256, 0, stream>>>(
        row_ptr, col_idx, scores, vbuf, osw);

    // out = attn_out @ Wo^T + bo (mode 2)
    wmma_gemm_kernel<<<dim3(HIDDEN / 128, NNODES / 128), 256, 0, stream>>>(
        osw, Wo_sw, HIDDEN / 16, bo, nullptr, out, nullptr, 2);
}
